// GCNClassifier_47502338294232
// MI455X (gfx1250) — matrix operand order repaired and checked by reading, not compiled, not run on hardware
//
#include <hip/hip_runtime.h>

#define NNODES 100000
#define NEDGES 1600000
#define DINF   128
#define HDIM   64
#define NCLS   10
#define NGRAPH 512

typedef __attribute__((ext_vector_type(16))) __bf16 v16bf;
typedef __attribute__((ext_vector_type(8)))  float  v8f;

// ---------------------------------------------------------------------------
// utility: fill float array with a constant
// ---------------------------------------------------------------------------
__global__ void fill_kernel(float* __restrict__ p, float v, int n) {
    int i = blockIdx.x * blockDim.x + threadIdx.x;
    if (i < n) p[i] = v;
}

// degree accumulation over explicit edges (self-loop pre-seeded as 1.0)
__global__ void deg_kernel(const int* __restrict__ dst, float* __restrict__ deg, int e) {
    int i = blockIdx.x * blockDim.x + threadIdx.x;
    if (i < e) atomicAdd(&deg[dst[i]], 1.0f);
}

// deg -> dinv in place (deg >= 1 always: self-loops guarantee it)
__global__ void dinv_kernel(float* __restrict__ d, int n) {
    int i = blockIdx.x * blockDim.x + threadIdx.x;
    if (i < n) d[i] = rsqrtf(d[i]);
}

// ---------------------------------------------------------------------------
// pack W[K,64] (f32, row-major) into bf16 hi/lo arrays pre-permuted into the
// kernel's own WMMA B lane order (contiguous k, same as A): packed[((kt*4+nt)*32 + lane)*16 + j] = W[k][n]
// with n = nt*16 + (lane&15), k = kt*32 + (lane>>4)*16 + j.
// Done once per call -> GEMM lanes do one contiguous 32B load per tile.
// ---------------------------------------------------------------------------
__global__ void pack_w_kernel(const float* __restrict__ W,
                              __bf16* __restrict__ hi,
                              __bf16* __restrict__ lo, int K) {
    int idx = blockIdx.x * blockDim.x + threadIdx.x;
    if (idx >= K * HDIM) return;
    int j    = idx & 15;
    int lane = (idx >> 4) & 31;
    int nt   = (idx >> 9) & 3;
    int kt   = idx >> 11;
    int n = nt * 16 + (lane & 15);
    int k = kt * 32 + (lane >> 4) * 16 + j;
    float f  = W[k * HDIM + n];
    __bf16 h = (__bf16)f;
    hi[idx] = h;
    lo[idx] = (__bf16)(f - (float)h);
}

// ---------------------------------------------------------------------------
// WMMA GEMM: Y[N,64] = X[N,K] @ W[K,64], f32 in/out, compensated-bf16 WMMA.
// Prepacked B staged in LDS once per block; one wave owns a 16x64 strip.
// ---------------------------------------------------------------------------
template <int K>
__global__ __launch_bounds__(256) void gemm_wmma(const float* __restrict__ X,
                                                 const __bf16* __restrict__ Whi,
                                                 const __bf16* __restrict__ Wlo,
                                                 float* __restrict__ Y,
                                                 int ntiles) {
    constexpr int BELEMS = K * HDIM;           // bf16 elements per array
    constexpr int BVECS  = BELEMS / 8;         // uint4 (16B) chunks
    __shared__ __align__(32) uint4 sHi4[BVECS];
    __shared__ __align__(32) uint4 sLo4[BVECS];

    // cooperative staging of packed B into LDS (all threads participate)
    {
        const uint4* gHi = (const uint4*)Whi;
        const uint4* gLo = (const uint4*)Wlo;
        for (int i = threadIdx.x; i < BVECS; i += 256) {
            sHi4[i] = gHi[i];
            sLo4[i] = gLo[i];
        }
    }
    __syncthreads();

    const int lane = threadIdx.x & 31;
    const int wave = threadIdx.x >> 5;
    const int tile = blockIdx.x * 8 + wave;
    if (tile < ntiles) {                      // wave-uniform guard: EXEC all-1s
        const __bf16* sBhi = (const __bf16*)sHi4;
        const __bf16* sBlo = (const __bf16*)sLo4;

        const int m0  = tile * 16;
        const int row = m0 + (lane & 15);
        const int kb  = (lane >> 4) * 16;     // A-matrix K sub-offset (0 or 16)
        const int col = lane & 15;            // C/D column

        v8f acc[4];
#pragma unroll
        for (int t = 0; t < 4; ++t)
#pragma unroll
            for (int r = 0; r < 8; ++r) acc[t][r] = 0.0f;

        for (int kt = 0; kt < K / 32; ++kt) {
            // ---- load A tile row (16 f32), split into bf16 hi/lo ----
            float af[16];
            const float* xp = X + (long long)row * K + kt * 32;
#pragma unroll
            for (int j = 0; j < 8; ++j) af[j]     = xp[kb + j];
#pragma unroll
            for (int j = 0; j < 8; ++j) af[8 + j] = xp[kb + 8 + j];

            v16bf ahi, alo;
#pragma unroll
            for (int j = 0; j < 16; ++j) {
                __bf16 h = (__bf16)af[j];
                ahi[j] = h;
                alo[j] = (__bf16)(af[j] - (float)h);
            }

#pragma unroll
            for (int nt = 0; nt < 4; ++nt) {
                const int boff = ((kt * 4 + nt) * 32 + lane) * 16;
                const v16bf bhi = *(const v16bf*)(sBhi + boff);
                const v16bf blo = *(const v16bf*)(sBlo + boff);
                // compensated product: hi*hi + lo*hi + hi*lo (f32 accumulate)
                acc[nt] = __builtin_amdgcn_wmma_f32_16x16x32_bf16(
                    false, ahi, false, bhi, (short)0, acc[nt], false, false);
                acc[nt] = __builtin_amdgcn_wmma_f32_16x16x32_bf16(
                    false, alo, false, bhi, (short)0, acc[nt], false, false);
                acc[nt] = __builtin_amdgcn_wmma_f32_16x16x32_bf16(
                    false, ahi, false, blo, (short)0, acc[nt], false, false);
            }
        }

        // ---- store D: lane<16 -> rows m0+0..7, lane>=16 -> rows m0+8..15 ----
        const int rbase = m0 + (lane >> 4) * 8;
#pragma unroll
        for (int nt = 0; nt < 4; ++nt)
#pragma unroll
            for (int r = 0; r < 8; ++r)
                Y[(long long)(rbase + r) * HDIM + nt * 16 + col] = acc[nt][r];
    }
}

// ---------------------------------------------------------------------------
// edge message pass: agg[dst] += dinv[src]*dinv[dst] * Y[src]  (64 feats/edge)
// one work item = (edge, 4-float chunk); atomics land in L2 (25.6MB target)
// ---------------------------------------------------------------------------
__global__ void scatter_kernel(const float* __restrict__ Y,
                               const int* __restrict__ src,
                               const int* __restrict__ dst,
                               const float* __restrict__ dinv,
                               float* __restrict__ agg) {
    const long long total = (long long)NEDGES * 16;
    for (long long idx = (long long)blockIdx.x * blockDim.x + threadIdx.x;
         idx < total; idx += (long long)gridDim.x * blockDim.x) {
        const int e = (int)(idx >> 4);
        const int c = ((int)idx & 15) * 4;
        const int s = src[e];
        const int d = dst[e];
        const float w = dinv[s] * dinv[d];
        const float4 v = *(const float4*)(Y + (long long)s * HDIM + c);
        float* ap = agg + (long long)d * HDIM + c;
        atomicAdd(ap + 0, w * v.x);
        atomicAdd(ap + 1, w * v.y);
        atomicAdd(ap + 2, w * v.z);
        atomicAdd(ap + 3, w * v.w);
    }
}

// epilogue layer 1: h = relu(agg + dinv^2 * XW + b), written in place over agg
__global__ void post_relu_kernel(float* __restrict__ agg,
                                 const float* __restrict__ Y,
                                 const float* __restrict__ dinv,
                                 const float* __restrict__ b) {
    int idx = blockIdx.x * blockDim.x + threadIdx.x;
    if (idx >= NNODES * HDIM) return;
    int i = idx >> 6, f = idx & 63;
    float di = dinv[i];
    float v = agg[idx] + di * di * Y[idx] + b[f];
    agg[idx] = v > 0.0f ? v : 0.0f;
}

// epilogue layer 2 fused with pooling: pooled[batch[i]] += relu(...), counts++
__global__ void post_pool_kernel(const float* __restrict__ agg,
                                 const float* __restrict__ Y,
                                 const float* __restrict__ dinv,
                                 const float* __restrict__ b,
                                 const int* __restrict__ batch,
                                 float* __restrict__ pooled,
                                 float* __restrict__ counts) {
    int idx = blockIdx.x * blockDim.x + threadIdx.x;
    if (idx >= NNODES * HDIM) return;
    int i = idx >> 6, f = idx & 63;
    float di = dinv[i];
    float v = agg[idx] + di * di * Y[idx] + b[f];
    v = v > 0.0f ? v : 0.0f;
    int g = batch[i];
    atomicAdd(&pooled[g * HDIM + f], v);
    if (f == 0) atomicAdd(&counts[g], 1.0f);
}

// classifier head: out[g,c] = (pooled[g]/max(cnt,1)) . Wl[:,c] + bl[c]
__global__ void final_kernel(const float* __restrict__ pooled,
                             const float* __restrict__ counts,
                             const float* __restrict__ Wl,
                             const float* __restrict__ bl,
                             float* __restrict__ out) {
    int idx = blockIdx.x * blockDim.x + threadIdx.x;
    if (idx >= NGRAPH * NCLS) return;
    int g = idx / NCLS, c = idx % NCLS;
    float inv = 1.0f / fmaxf(counts[g], 1.0f);
    float s = bl[c];
#pragma unroll 8
    for (int f = 0; f < HDIM; ++f)
        s += pooled[g * HDIM + f] * inv * Wl[f * NCLS + c];
    out[idx] = s;
}

// ---------------------------------------------------------------------------
extern "C" void kernel_launch(void* const* d_in, const int* in_sizes, int n_in,
                              void* d_out, int out_size, void* d_ws, size_t ws_size,
                              hipStream_t stream) {
    (void)in_sizes; (void)n_in; (void)out_size; (void)ws_size;

    const float* x     = (const float*)d_in[0];
    const int*   ei    = (const int*)d_in[1];
    const int*   batch = (const int*)d_in[2];
    const float* W1    = (const float*)d_in[3];
    const float* b1    = (const float*)d_in[4];
    const float* W2    = (const float*)d_in[5];
    const float* b2    = (const float*)d_in[6];
    const float* Wl    = (const float*)d_in[7];
    const float* bl    = (const float*)d_in[8];
    float*       out   = (float*)d_out;

    const int* src = ei;            // edge_index[0]
    const int* dst = ei + NEDGES;   // edge_index[1]

    float* wsf    = (float*)d_ws;
    float* dinv   = wsf;                                   // N
    float* bufA   = dinv + NNODES;                         // N*64  (XW1 -> HW2)
    float* bufB   = bufA + (size_t)NNODES * HDIM;          // N*64  (agg1/h1 -> agg2)
    float* pooled = bufB + (size_t)NNODES * HDIM;          // G*64
    float* counts = pooled + NGRAPH * HDIM;                // G
    // bf16 prepacked weights (region is 32B aligned: preceding floats are)
    __bf16* Whi1 = (__bf16*)(counts + NGRAPH);             // 128*64
    __bf16* Wlo1 = Whi1 + DINF * HDIM;                     // 128*64
    __bf16* Whi2 = Wlo1 + DINF * HDIM;                     // 64*64
    __bf16* Wlo2 = Whi2 + HDIM * HDIM;                     // 64*64

    const int T = 256;
    const int tiles   = NNODES / 16;        // 6250, exact
    const int gblocks = (tiles + 7) / 8;    // 782 blocks x 8 waves

    // --- init (deterministic: redone every call) ---
    fill_kernel<<<(NNODES + T - 1) / T, T, 0, stream>>>(dinv, 1.0f, NNODES);  // self-loop deg
    fill_kernel<<<(NNODES * HDIM + T - 1) / T, T, 0, stream>>>(bufB, 0.0f, NNODES * HDIM);
    fill_kernel<<<(NGRAPH * HDIM + NGRAPH + T - 1) / T, T, 0, stream>>>(pooled, 0.0f,
                                                                        NGRAPH * HDIM + NGRAPH);
    // --- weight prepack (bf16 hi/lo in WMMA B layout) ---
    pack_w_kernel<<<(DINF * HDIM + T - 1) / T, T, 0, stream>>>(W1, Whi1, Wlo1, DINF);
    pack_w_kernel<<<(HDIM * HDIM + T - 1) / T, T, 0, stream>>>(W2, Whi2, Wlo2, HDIM);

    // --- degree / norm ---
    deg_kernel<<<(NEDGES + T - 1) / T, T, 0, stream>>>(dst, dinv, NEDGES);
    dinv_kernel<<<(NNODES + T - 1) / T, T, 0, stream>>>(dinv, NNODES);

    // --- layer 1 ---
    gemm_wmma<DINF><<<gblocks, 256, 0, stream>>>(x, Whi1, Wlo1, bufA, tiles);
    scatter_kernel<<<8192, 256, 0, stream>>>(bufA, src, dst, dinv, bufB);
    post_relu_kernel<<<(NNODES * HDIM + T - 1) / T, T, 0, stream>>>(bufB, bufA, dinv, b1);

    // --- layer 2 (pooling fused into epilogue) ---
    gemm_wmma<HDIM><<<gblocks, 256, 0, stream>>>(bufB, Whi2, Wlo2, bufA, tiles);
    fill_kernel<<<(NNODES * HDIM + T - 1) / T, T, 0, stream>>>(bufB, 0.0f, NNODES * HDIM);
    scatter_kernel<<<8192, 256, 0, stream>>>(bufA, src, dst, dinv, bufB);
    post_pool_kernel<<<(NNODES * HDIM + T - 1) / T, T, 0, stream>>>(bufB, bufA, dinv, b2,
                                                                    batch, pooled, counts);

    // --- classifier head ---
    final_kernel<<<(NGRAPH * NCLS + T - 1) / T, T, 0, stream>>>(pooled, counts, Wl, bl, out);
}
